// ImplicitMap_fExtrator_70918499991664
// MI455X (gfx1250) — matrix operand order repaired and checked by reading, not compiled, not run on hardware
//
#include <hip/hip_runtime.h>
#include <hip/hip_bf16.h>

// ---------------------------------------------------------------------------
// Fused 12-layer implicit-MLP + forward-mode xyz gradient for MI455X (gfx1250).
//
// Gradient is seeded with identity basis columns, so every layer is ONE GEMM
// over 4 augmented rows per point: [x; gx; gy; gz]; then per point:
//   z-row: softplus(100z)/100 (+bias), g-rows: sigmoid(100z) * row.
// Compute-bound (~0.7 TFLOP vs ~40 MB HBM) -> bf16 WMMA, f32 accumulate.
//
// Fast path: weights pre-padded (32-multiples) + pre-converted to bf16 in d_ws
// once per launch -> the k-loop is 2x ds_load_b128 + 4x global_load_b128 +
// 2x v_wmma (two independent accumulator chains sharing one A fragment).
// ---------------------------------------------------------------------------

typedef __attribute__((ext_vector_type(16))) __bf16 v16bf;
typedef __attribute__((ext_vector_type(8)))  __bf16 v8bf;
typedef __attribute__((ext_vector_type(8)))  float  v8f;

#define NPTS        16384      // 2 * 8192 points
#define NLAYERS     12
#define ROWS        64         // 16 points * 4 augmented rows per workgroup
#define LDS_STRIDE  976        // bf16 elems per LDS row (max 960 cols + pad)

__device__ __forceinline__ float sp100(float z) {
    return fmaxf(z, 0.0f) + 0.01f * log1pf(__expf(-100.0f * fabsf(z)));
}
__device__ __forceinline__ float sig100(float z) {
    return 1.0f / (1.0f + __expf(-100.0f * z));
}

// A fragment (16x32 activation tile) from LDS, contiguous K order matching B:
// elements 0..15 = K(16*half .. +15).
__device__ __forceinline__ v16bf load_afrag(const __bf16* Xrow, int kb, int half) {
    v8bf lo = *(const v8bf*)(Xrow + kb + 16 * half);
    v8bf hi = *(const v8bf*)(Xrow + kb + 16 * half + 8);
    return __builtin_shufflevector(lo, hi, 0, 1, 2, 3, 4, 5, 6, 7,
                                           8, 9, 10, 11, 12, 13, 14, 15);
}

// B fragment from padded bf16 weights: 16 contiguous K values per lane.
__device__ __forceinline__ v16bf load_bfrag(const __bf16* wp) {
    v8bf lo = *(const v8bf*)(wp);
    v8bf hi = *(const v8bf*)(wp + 8);
    return __builtin_shufflevector(lo, hi, 0, 1, 2, 3, 4, 5, 6, 7,
                                           8, 9, 10, 11, 12, 13, 14, 15);
}

// =====================  weight pre-pad / pre-convert  ======================
__global__ void pad_weights_kernel(const float* __restrict__ W,
                                   const float* __restrict__ Bsrc,
                                   __bf16* __restrict__ Wp,
                                   float* __restrict__ Bp,
                                   int cout, int cin, int rpad, int cpad) {
    const int total = rpad * cpad;
    for (int i = blockIdx.x * blockDim.x + threadIdx.x; i < total;
         i += gridDim.x * blockDim.x) {
        int r = i / cpad, c = i - r * cpad;
        float v = (r < cout && c < cin) ? W[(size_t)r * cin + c] : 0.0f;
        Wp[i] = (__bf16)v;
    }
    for (int i = blockIdx.x * blockDim.x + threadIdx.x; i < rpad;
         i += gridDim.x * blockDim.x)
        Bp[i] = (i < cout) ? Bsrc[i] : 0.0f;
}

// ============================  fast MLP kernel  ============================
struct ParamsFast {
    const float*  input;    // (2, 8192, 3)
    const float*  latent;   // (2, 256)
    const __bf16* Wp[12];   // padded bf16 weights (rpad x cpad)
    const float*  Bp[12];   // padded f32 biases (rpad)
    float*        out;      // z | g | input_con
};

__global__ void __launch_bounds__(256, 1)
ImplicitMap_fExtrator_70918499991664_kernel(ParamsFast P) {
    constexpr int cins[12]  = {259, 515, 512, 512, 576, 576, 768, 768, 768, 960, 960, 896};
    constexpr int couts[12] = {515, 512, 512, 576, 576, 768, 768, 768, 960, 960, 896, 1};

    extern __shared__ __align__(16) char smem_raw[];   // 2 * 64 * 976 * 2 B
    __bf16* buf0 = (__bf16*)smem_raw;
    __bf16* buf1 = buf0 + ROWS * LDS_STRIDE;

    const int t      = threadIdx.x;
    const int wg     = blockIdx.x;        // 16 points per workgroup
    const int w      = t >> 5;            // wave 0..7 (wave32)
    const int stripe = w & 3;             // which 16 augmented rows
    const int osplit = w >> 2;            // wave-pair splits output tiles
    const int lane   = t & 31;
    const int lr     = lane & 15;
    const int half   = lane >> 4;

    // ---- prologue: stage [x; ex; ey; ez] rows (bf16) + emit input_con ----
    for (int idx = t; idx < 16 * 288; idx += 256) {        // 288 = ceil32(259)
        int lp = idx / 288, c = idx - 288 * lp;
        int p = wg * 16 + lp;
        int b = p >> 13, n = p & 8191;
        float xv = 0.0f;
        if (c < 3)        xv = P.input[((size_t)b * 8192 + n) * 3 + c];
        else if (c < 259) xv = P.latent[b * 256 + (c - 3)];
        __bf16* row0 = buf0 + (4 * lp) * LDS_STRIDE;
        row0[c]                  = (__bf16)xv;
        row0[1 * LDS_STRIDE + c] = (__bf16)((c == 0) ? 1.0f : 0.0f);
        row0[2 * LDS_STRIDE + c] = (__bf16)((c == 1) ? 1.0f : 0.0f);
        row0[3 * LDS_STRIDE + c] = (__bf16)((c == 2) ? 1.0f : 0.0f);
        if (c < 259) P.out[65536 + ((size_t)b * 8192 + n) * 259 + c] = xv;
    }

    for (int l = 0; l < NLAYERS; ++l) {
        __syncthreads();
        const int cin    = cins[l], cout = couts[l];
        const int cpad   = (cin + 31) & ~31;
        const int rpad   = (cout + 31) & ~31;
        const int ktiles = cpad >> 5;
        const int otiles = rpad >> 4;               // always even
        const __bf16* Xin  = (l & 1) ? buf1 : buf0;
        __bf16*       Xout = (l & 1) ? buf0 : buf1;
        const __bf16* Wl   = P.Wp[l];
        const float*  Bl   = P.Bp[l];
        const __bf16* Arow = Xin + (16 * stripe + lr) * LDS_STRIDE;
        const bool last    = (l == NLAYERS - 1);

        // wave-pairs interleave over pairs of 16-wide output tiles
        for (int oc = osplit * 2; oc < otiles; oc += 4) {
            const int ob0 = oc * 16;
            const __bf16* w0 = Wl + (size_t)(ob0 + lr) * cpad + 16 * half;
            const __bf16* w1 = w0 + (size_t)16 * cpad;

            v8f a0 = {}, a1 = {};
            for (int kt = 0; kt < ktiles; ++kt) {
                const int kb = kt * 32;
                v16bf a  = load_afrag(Arow, kb, half);
                v16bf b0 = load_bfrag(w0 + kb);
                v16bf b1 = load_bfrag(w1 + kb);
                if (kt + 2 < ktiles) __builtin_prefetch(w0 + kb + 64, 0, 1);
                // two independent f32 accumulation chains share one A fragment
                a0 = __builtin_amdgcn_wmma_f32_16x16x32_bf16(
                         false, a, false, b0, (short)0, a0, false, false);
                a1 = __builtin_amdgcn_wmma_f32_16x16x32_bf16(
                         false, a, false, b1, (short)0, a1, false, false);
            }

            if (!last) {
#pragma unroll
                for (int u = 0; u < 2; ++u) {
                    v8f  acc   = u ? a1 : a0;
                    int  col   = ob0 + u * 16 + lr;
                    bool valid = (col < cout);
                    float bv = Bl[col];                 // padded: in-bounds, 0 pad
                    float z0 = acc[0] + bv, z4 = acc[4] + bv;
                    float x0 = sp100(z0), x4 = sp100(z4);
                    float s0 = sig100(z0), s4 = sig100(z4);
                    float vals[8] = { x0, acc[1] * s0, acc[2] * s0, acc[3] * s0,
                                      x4, acc[5] * s4, acc[6] * s4, acc[7] * s4 };
#pragma unroll
                    for (int r = 0; r < 8; ++r) {
                        float v = valid ? vals[r] : 0.0f;   // pad cols exactly 0
                        Xout[(16 * stripe + r + 8 * half) * LDS_STRIDE + col] =
                            (__bf16)v;
                    }
                }
            } else if (lr == 0 && oc == 0) {            // cout == 1: tile 0, col 0
                float bv = Bl[0];
                float g[8] = { a0[0] + bv, a0[1], a0[2], a0[3],
                               a0[4] + bv, a0[5], a0[6], a0[7] };
#pragma unroll
                for (int r = 0; r < 8; ++r) {
                    int rr = r + 8 * half;
                    int s  = rr & 3;                    // 0 = z, 1..3 = grad comps
                    int p  = wg * 16 + 4 * stripe + (rr >> 2);
                    if (s == 0) P.out[p] = g[r];                          // (B,1,N)
                    else        P.out[16384 + (size_t)p * 3 + (s - 1)] = g[r];
                }
            }
        }
    }
}

// ====================  fallback (convert-in-flight)  =======================
struct ParamsFb {
    const float* input;
    const float* latent;
    const float* W[12];
    const float* B[12];
    float*       out;
};

__device__ __forceinline__ v16bf load_bfrag_guard(const float* Wl, int cout, int cin,
                                                  int row, int colbase) {
    const bool rok = row < cout;
    const float* wp = Wl + (size_t)(rok ? row : (cout - 1)) * cin;
    v16bf r;
#pragma unroll
    for (int e = 0; e < 16; ++e) {
        int c  = colbase + e;
        int cc = (c < cin) ? c : (cin - 1);
        float f = wp[cc];
        r[e] = (__bf16)((rok && c < cin) ? f : 0.0f);
    }
    return r;
}

__global__ void __launch_bounds__(128, 1)
implicit_mlp_fallback(ParamsFb P) {
    constexpr int cins[12]  = {259, 515, 512, 512, 576, 576, 768, 768, 768, 960, 960, 896};
    constexpr int couts[12] = {515, 512, 512, 576, 576, 768, 768, 768, 960, 960, 896, 1};

    extern __shared__ __align__(16) char smem_raw[];
    __bf16* buf0 = (__bf16*)smem_raw;
    __bf16* buf1 = buf0 + ROWS * LDS_STRIDE;

    const int t = threadIdx.x, wg = blockIdx.x;
    const int w = t >> 5, lane = t & 31, lr = lane & 15, half = lane >> 4;

    for (int idx = t; idx < 16 * 288; idx += 128) {
        int lp = idx / 288, c = idx - 288 * lp;
        int p = wg * 16 + lp;
        int b = p >> 13, n = p & 8191;
        float xv = 0.0f;
        if (c < 3)        xv = P.input[((size_t)b * 8192 + n) * 3 + c];
        else if (c < 259) xv = P.latent[b * 256 + (c - 3)];
        __bf16* row0 = buf0 + (4 * lp) * LDS_STRIDE;
        row0[c]                  = (__bf16)xv;
        row0[1 * LDS_STRIDE + c] = (__bf16)((c == 0) ? 1.0f : 0.0f);
        row0[2 * LDS_STRIDE + c] = (__bf16)((c == 1) ? 1.0f : 0.0f);
        row0[3 * LDS_STRIDE + c] = (__bf16)((c == 2) ? 1.0f : 0.0f);
        if (c < 259) P.out[65536 + ((size_t)b * 8192 + n) * 259 + c] = xv;
    }

    for (int l = 0; l < NLAYERS; ++l) {
        __syncthreads();
        const int cin = cins[l], cout = couts[l];
        const int ktiles = (cin + 31) >> 5;
        const int otiles = ((cout + 31) & ~31) >> 4;
        const __bf16* Xin  = (l & 1) ? buf1 : buf0;
        __bf16*       Xout = (l & 1) ? buf0 : buf1;
        const float* Wl = P.W[l];
        const float* Bl = P.B[l];
        const __bf16* Arow = Xin + (16 * w + lr) * LDS_STRIDE;
        const bool last = (l == NLAYERS - 1);
        const int o_end = last ? 1 : otiles;

        for (int ot = 0; ot < o_end; ++ot) {
            const int ob = ot * 16, col = ob + lr;
            if (ob >= cout) {
#pragma unroll
                for (int r = 0; r < 8; ++r)
                    Xout[(16 * w + r + 8 * half) * LDS_STRIDE + col] = (__bf16)0.0f;
                continue;
            }
            v8f acc = {};
            for (int kt = 0; kt < ktiles; ++kt) {
                const int kb = kt * 32;
                v16bf a = load_afrag(Arow, kb, half);
                v16bf b = load_bfrag_guard(Wl, cout, cin, col, kb + 16 * half);
                acc = __builtin_amdgcn_wmma_f32_16x16x32_bf16(
                          false, a, false, b, (short)0, acc, false, false);
            }
            bool valid = (col < cout);
            float bv = valid ? Bl[col] : 0.0f;
            float z0 = acc[0] + bv, z4 = acc[4] + bv;
            if (!last) {
                float x0 = sp100(z0), x4 = sp100(z4);
                float s0 = sig100(z0), s4 = sig100(z4);
                float vals[8] = { x0, acc[1] * s0, acc[2] * s0, acc[3] * s0,
                                  x4, acc[5] * s4, acc[6] * s4, acc[7] * s4 };
#pragma unroll
                for (int r = 0; r < 8; ++r) {
                    float v = valid ? vals[r] : 0.0f;
                    Xout[(16 * w + r + 8 * half) * LDS_STRIDE + col] = (__bf16)v;
                }
            } else if (lr == 0) {
                float g[8] = { z0, acc[1], acc[2], acc[3], z4, acc[5], acc[6], acc[7] };
#pragma unroll
                for (int r = 0; r < 8; ++r) {
                    int rr = r + 8 * half, s = rr & 3;
                    int p = wg * 16 + 4 * w + (rr >> 2);
                    if (s == 0) P.out[p] = g[r];
                    else        P.out[16384 + (size_t)p * 3 + (s - 1)] = g[r];
                }
            }
        }
    }
}

// ==============================  launcher  =================================
extern "C" void kernel_launch(void* const* d_in, const int* in_sizes, int n_in,
                              void* d_out, int out_size, void* d_ws, size_t ws_size,
                              hipStream_t stream) {
    (void)in_sizes; (void)n_in; (void)out_size;
    static const int cins[12]  = {259, 515, 512, 512, 576, 576, 768, 768, 768, 960, 960, 896};
    static const int couts[12] = {515, 512, 512, 576, 576, 768, 768, 768, 960, 960, 896, 1};

    // workspace layout: [bf16 padded weights][f32 padded biases]
    size_t wOff[12], bOff[12], wTot = 0, bTot = 0;
    for (int l = 0; l < 12; ++l) {
        int cpad = (cins[l] + 31) & ~31, rpad = (couts[l] + 31) & ~31;
        wOff[l] = wTot;  wTot += (size_t)rpad * cpad;
        bOff[l] = bTot;  bTot += (size_t)rpad;
    }
    const size_t wBytes = (wTot * 2 + 15) & ~(size_t)15;
    const size_t need   = wBytes + bTot * 4;

    const size_t smem = (size_t)2 * ROWS * LDS_STRIDE * 2;  // 249,856 B LDS/WGP
    const int blocks = NPTS / 16;                           // 1024 workgroups

    if (d_ws != nullptr && ws_size >= need) {
        __bf16* wsW = (__bf16*)d_ws;
        float*  wsB = (float*)((char*)d_ws + wBytes);
        ParamsFast P;
        P.input  = (const float*)d_in[0];
        P.latent = (const float*)d_in[1];
        for (int l = 0; l < 12; ++l) {
            const float* W = (const float*)d_in[2 + 2 * l];
            const float* B = (const float*)d_in[3 + 2 * l];
            int cpad = (cins[l] + 31) & ~31, rpad = (couts[l] + 31) & ~31;
            int total = rpad * cpad;
            int grd = (total + 255) / 256; if (grd > 1024) grd = 1024;
            pad_weights_kernel<<<grd, 256, 0, stream>>>(
                W, B, wsW + wOff[l], wsB + bOff[l], couts[l], cins[l], rpad, cpad);
            P.Wp[l] = wsW + wOff[l];
            P.Bp[l] = wsB + bOff[l];
        }
        P.out = (float*)d_out;
        ImplicitMap_fExtrator_70918499991664_kernel<<<blocks, 256, smem, stream>>>(P);
    } else {
        ParamsFb P;
        P.input  = (const float*)d_in[0];
        P.latent = (const float*)d_in[1];
        for (int l = 0; l < 12; ++l) {
            P.W[l] = (const float*)d_in[2 + 2 * l];
            P.B[l] = (const float*)d_in[3 + 2 * l];
        }
        P.out = (float*)d_out;
        implicit_mlp_fallback<<<blocks, 128, smem, stream>>>(P);
    }
}
